// CARP3D_LD_Ave_44994077393523
// MI455X (gfx1250) — matrix operand order repaired and checked by reading, not compiled, not run on hardware
//
#include <hip/hip_runtime.h>
#include <cmath>

typedef __attribute__((ext_vector_type(16))) _Float16 v16h;
typedef __attribute__((ext_vector_type(2)))  __fp16   v2fp;
typedef __attribute__((ext_vector_type(8)))  float    v8f;
typedef __attribute__((ext_vector_type(4)))  unsigned int u32x4;
typedef __attribute__((ext_vector_type(8)))  unsigned int u32x8;

#define B_BAGS 16
#define N_INST 8192
#define F_DIM  512
#define H_DIM  512
#define D_DIM  256
#define TM     32
#define ROWS_TOTAL (B_BAGS * N_INST)   // 131072

union AFrag { v16h v; unsigned int u[8]; };
union H2U   { v2fp h; unsigned int u; };

// A-matrix 16x32 f16 fragment from f16 LDS (row-major, `stride` f16/row).
// ISA 7.12.2: lane = M (l&15); lanes>=16 take K base 8; VGPR j holds a K pair.
__device__ inline v16h load_a_frag_f16(const _Float16* lds, int row0, int stride, int kbase) {
  const int l  = threadIdx.x & 31;
  const int m  = l & 15;
  const int kb = (l >> 4) << 3;
  AFrag f;
#pragma unroll
  for (int j = 0; j < 8; ++j) {
    const int k = kbase + kb + ((j & 3) << 1) + ((j >> 2) << 4);
    f.u[j] = *(const unsigned int*)&lds[(row0 + m) * stride + k];
  }
  return f.v;
}

// A-fragment from f32 LDS tile (stride 32 f32/row), packing pairs with v_cvt_pk_f16_f32.
__device__ inline v16h load_a_frag_f32(const float* lds, int row0) {
  const int l  = threadIdx.x & 31;
  const int m  = l & 15;
  const int kb = (l >> 4) << 3;
  AFrag f;
#pragma unroll
  for (int j = 0; j < 8; ++j) {
    const int k = kb + ((j & 3) << 1) + ((j >> 2) << 4);
    const float2 t = *(const float2*)&lds[(row0 + m) * 32 + k];
    H2U p; p.h = __builtin_amdgcn_cvt_pkrtz(t.x, t.y);
    f.u[j] = p.u;
  }
  return f.v;
}

__device__ inline v16h load_b_frag_f16(const _Float16* g, int stride) {
  const int l  = threadIdx.x & 31;
  const int kb = (l >> 4) << 3;
  v16h b;
#pragma unroll
  for (int s = 0; s < 16; ++s) {
    const int k = kb + (s & 7) + ((s >> 3) << 4);
    b[s] = g[(size_t)k * stride + (l & 15)];
  }
  return b;
}
// ---------------- prep: fp32 -> f16 weight conversion ----------------
__global__ void cvt_f32_f16(const float* __restrict__ src, _Float16* __restrict__ dst, int n) {
  int i = blockIdx.x * blockDim.x + threadIdx.x;
  if (i < n) dst[i] = (_Float16)src[i];
}

// ---------------- phase 1: fused h1 + attention scores ----------------
__global__ __launch_bounds__(256)
void mil_phase1(const float* __restrict__ h,
                const _Float16* __restrict__ W1h,
                const float* __restrict__ b1,
                const _Float16* __restrict__ Wah,
                const float* __restrict__ ba,
                const _Float16* __restrict__ Wbh,
                const float* __restrict__ bb,
                const float* __restrict__ Wc,
                const float* __restrict__ bc,
                _Float16* __restrict__ h1h,
                float* __restrict__ scores,
                float* __restrict__ a_raw_out)
{
  __shared__ float    sA32[TM * 32];        // 4KB: TDM-staged h chunk (f32)
  __shared__ _Float16 sH1[TM * H_DIM];      // 32KB: h1 tile (f16)
  __shared__ float    sScore[TM];

  const int tid  = threadIdx.x;
  const int wave = tid >> 5;
  const int lane = tid & 31;
  const int rowbase = blockIdx.x * TM;

  // ---- Step A: h1_tile = relu(h_tile @ W1 + b1), [TM x H] ----
  const int sr = wave & 1;       // row strip (16 rows)
  const int cg = wave >> 1;      // column group: cols cg*128 .. +127 (8 tiles)
  v8f acc[8] = {};

  // D# group 1 (invariant): data_size=4B, 32x32 tile, tensor dims 32x32, dim0 stride=512
  u32x8 g1;
  g1[0] = (2u << 16);            // data_size = 4 bytes
  g1[1] = (32u << 16);           // tensor_dim0[15:0] into bits 63:48
  g1[2] = (32u << 16);           // tensor_dim1[15:0] into bits 95:80
  g1[3] = (32u << 16);           // tile_dim0 into bits 127:112
  g1[4] = 32u;                   // tile_dim1 = 32, tile_dim2 = 0
  g1[5] = (unsigned)F_DIM;       // tensor_dim0_stride = 512 (low 32 of 48)
  g1[6] = 0u;
  g1[7] = 0u;
  const unsigned lds_stage = (unsigned)(size_t)(void*)&sA32[0];

  for (int kc = 0; kc < F_DIM / 32; ++kc) {
    __syncthreads();
    if (wave == 0) {
      // D# group 0: count=1, lds_addr, global tile address, type=2 ("image")
      const unsigned long long ga =
          (unsigned long long)(const void*)(h + (size_t)rowbase * F_DIM + kc * 32);
      u32x4 g0;
      g0[0] = 1u;
      g0[1] = lds_stage;
      g0[2] = (unsigned)ga;
      g0[3] = (unsigned)((ga >> 32) & 0x01FFFFFFull) | (2u << 30);
      asm volatile("tensor_load_to_lds %0, %1" :: "s"(g0), "s"(g1) : "memory");
      __builtin_amdgcn_s_wait_tensorcnt(0);
    }
    __syncthreads();

    const v16h afrag = load_a_frag_f32(sA32, sr * 16);
    const _Float16* wrow = &W1h[(size_t)(kc * 32) * H_DIM + cg * 128];
#pragma unroll
    for (int hf = 0; hf < 2; ++hf) {   // two half-batches of 4 tiles: fewer live B regs
      const v16h b0 = load_b_frag_f16(&wrow[hf * 64 +  0], H_DIM);
      const v16h b1 = load_b_frag_f16(&wrow[hf * 64 + 16], H_DIM);
      const v16h b2 = load_b_frag_f16(&wrow[hf * 64 + 32], H_DIM);
      const v16h b3 = load_b_frag_f16(&wrow[hf * 64 + 48], H_DIM);
      acc[hf*4+0] = __builtin_amdgcn_wmma_f32_16x16x32_f16(false, afrag, false, b0, (short)0, acc[hf*4+0], false, false);
      acc[hf*4+1] = __builtin_amdgcn_wmma_f32_16x16x32_f16(false, afrag, false, b1, (short)0, acc[hf*4+1], false, false);
      acc[hf*4+2] = __builtin_amdgcn_wmma_f32_16x16x32_f16(false, afrag, false, b2, (short)0, acc[hf*4+2], false, false);
      acc[hf*4+3] = __builtin_amdgcn_wmma_f32_16x16x32_f16(false, afrag, false, b3, (short)0, acc[hf*4+3], false, false);
    }
  }

  // bias + relu -> sH1 (f16)
#pragma unroll
  for (int ct = 0; ct < 8; ++ct) {
    const int n = cg * 128 + ct * 16 + (lane & 15);
    const int mbase = sr * 16 + ((lane >> 4) << 3);
    const float bias = b1[n];
#pragma unroll
    for (int r = 0; r < 8; ++r) {
      float v = acc[ct][r] + bias;
      v = v > 0.f ? v : 0.f;
      sH1[(mbase + r) * H_DIM + n] = (_Float16)v;
    }
  }
  if (tid < TM) sScore[tid] = 0.f;
  __syncthreads();

  // persist h1 tile to workspace: async LDS -> global DMA (no VGPR round-trip)
  {
    const unsigned long long gdst0 =
        (unsigned long long)(void*)(h1h + (size_t)rowbase * H_DIM + tid * 8);
    const unsigned lsrc0 = (unsigned)(size_t)(void*)&sH1[tid * 8];
#pragma unroll
    for (int i = 0; i < 8; ++i) {
      const unsigned long long g = gdst0 + (unsigned long long)i * 4096ull; // 2048 f16 * 2B
      const unsigned ls = lsrc0 + i * 4096u;
      asm volatile("global_store_async_from_lds_b128 %0, %1, off"
                   :: "v"(g), "v"(ls) : "memory");
    }
  }

  // ---- Step B: a = tanh(h1@Wa+ba), g = sigmoid(h1@Wb+bb), score = (a*g)@Wc ----
#pragma unroll 1
  for (int p = 0; p < 4; ++p) {
    const int pos = wave * 4 + p;   // 0..31
    const int psr = pos & 1;
    const int dt  = pos >> 1;       // 0..15 (D tile)
    v8f accA = {};
    v8f accG = {};
    for (int kc = 0; kc < H_DIM / 32; ++kc) {
      const v16h afrag = load_a_frag_f16(sH1, psr * 16, H_DIM, kc * 32);
      const v16h bfa = load_b_frag_f16(&Wah[(size_t)(kc * 32) * D_DIM + dt * 16], D_DIM);
      const v16h bfb = load_b_frag_f16(&Wbh[(size_t)(kc * 32) * D_DIM + dt * 16], D_DIM);
      accA = __builtin_amdgcn_wmma_f32_16x16x32_f16(false, afrag, false, bfa, (short)0, accA, false, false);
      accG = __builtin_amdgcn_wmma_f32_16x16x32_f16(false, afrag, false, bfb, (short)0, accG, false, false);
    }
    const int d = dt * 16 + (lane & 15);
    const float bda = ba[d], bdb = bb[d], wc = Wc[d];
#pragma unroll
    for (int r = 0; r < 8; ++r) {
      const float a = tanhf(accA[r] + bda);
      const float g = 1.f / (1.f + expf(-(accG[r] + bdb)));
      float s = a * g * wc;
      s += __shfl_xor(s, 1, 16);
      s += __shfl_xor(s, 2, 16);
      s += __shfl_xor(s, 4, 16);
      s += __shfl_xor(s, 8, 16);
      if ((lane & 15) == 0) {
        const int m = psr * 16 + r + ((lane >> 4) << 3);
        atomicAdd(&sScore[m], s);
      }
    }
  }
  __syncthreads();
  if (tid < TM) {
    const float sc = sScore[tid] + bc[0];
    const int gi = rowbase + tid;
    scores[gi]    = sc;
    a_raw_out[gi] = sc;
  }
  // ensure async LDS->global h1 stores completed before wave exit
  asm volatile("s_wait_asynccnt 0x0" ::: "memory");
}

// ---------------- phase 2: per-bag softmax stats ----------------
__global__ __launch_bounds__(256)
void mil_phase2(const float* __restrict__ scores, float* __restrict__ stats) {
  __shared__ float red[256];
  const int b = blockIdx.x, tid = threadIdx.x;
  const float* s = scores + (size_t)b * N_INST;
  float mx = -3.4e38f;
  for (int n = tid; n < N_INST; n += 256) mx = fmaxf(mx, s[n]);
  red[tid] = mx; __syncthreads();
  for (int st = 128; st > 0; st >>= 1) {
    if (tid < st) red[tid] = fmaxf(red[tid], red[tid + st]);
    __syncthreads();
  }
  mx = red[0]; __syncthreads();
  float sm = 0.f;
  for (int n = tid; n < N_INST; n += 256) sm += expf(s[n] - mx);
  red[tid] = sm; __syncthreads();
  for (int st = 128; st > 0; st >>= 1) {
    if (tid < st) red[tid] += red[tid + st];
    __syncthreads();
  }
  if (tid == 0) { stats[b * 2] = mx; stats[b * 2 + 1] = red[0]; }
}

// ---------------- phase 3: M = softmax(A) @ h1 ----------------
__global__ __launch_bounds__(256)
void mil_phase3(const _Float16* __restrict__ h1h, const float* __restrict__ scores,
                const float* __restrict__ stats, float* __restrict__ m_out) {
  __shared__ float part[256];
  const int b   = blockIdx.x >> 2;
  const int cgp = blockIdx.x & 3;
  const int tid = threadIdx.x;
  const int col = cgp * 128 + (tid & 127);
  const int sl  = tid >> 7;
  const float mx  = stats[b * 2];
  const float inv = 1.f / stats[b * 2 + 1];
  const float* s = scores + (size_t)b * N_INST;
  const _Float16* hp = h1h + (size_t)b * N_INST * H_DIM;
  float acc = 0.f;
  for (int n = sl; n < N_INST; n += 2) {
    const float w = expf(s[n] - mx);
    acc += w * (float)hp[(size_t)n * H_DIM + col];
  }
  part[tid] = acc; __syncthreads();
  if (sl == 0)
    m_out[b * H_DIM + col] = (part[tid] + part[tid + 128]) * inv;
}

// ---------------- phase 4: mean over bags + classifier ----------------
__global__ __launch_bounds__(256)
void mil_phase4(const float* __restrict__ M, const float* __restrict__ Wcls,
                const float* __restrict__ bcls, float* __restrict__ out5) {
  __shared__ float r0[256], r1[256];
  const int tid = threadIdx.x;
  float p0 = 0.f, p1 = 0.f;
  for (int hh = tid; hh < H_DIM; hh += 256) {
    float ctx = 0.f;
    for (int b = 0; b < B_BAGS; ++b) ctx += M[b * H_DIM + hh];
    ctx *= (1.f / B_BAGS);
    p0 += ctx * Wcls[hh * 2 + 0];
    p1 += ctx * Wcls[hh * 2 + 1];
  }
  r0[tid] = p0; r1[tid] = p1; __syncthreads();
  for (int st = 128; st > 0; st >>= 1) {
    if (tid < st) { r0[tid] += r0[tid + st]; r1[tid] += r1[tid + st]; }
    __syncthreads();
  }
  if (tid == 0) {
    const float l0 = r0[0] + bcls[0], l1 = r1[0] + bcls[1];
    const float m  = fmaxf(l0, l1);
    const float e0 = expf(l0 - m), e1 = expf(l1 - m);
    const float inv = 1.f / (e0 + e1);
    out5[0] = l0; out5[1] = l1;
    out5[2] = e0 * inv; out5[3] = e1 * inv;
    out5[4] = (l1 > l0) ? 1.0f : 0.0f;
  }
}

extern "C" void kernel_launch(void* const* d_in, const int* in_sizes, int n_in,
                              void* d_out, int out_size, void* d_ws, size_t ws_size,
                              hipStream_t stream) {
  (void)in_sizes; (void)n_in; (void)out_size; (void)ws_size;
  const float* h    = (const float*)d_in[0];
  const float* W1   = (const float*)d_in[1];
  const float* b1   = (const float*)d_in[2];
  const float* Wa   = (const float*)d_in[3];
  const float* ba   = (const float*)d_in[4];
  const float* Wb   = (const float*)d_in[5];
  const float* bb   = (const float*)d_in[6];
  const float* Wc   = (const float*)d_in[7];
  const float* bc   = (const float*)d_in[8];
  const float* Wcls = (const float*)d_in[9];
  const float* bcls = (const float*)d_in[10];
  float* out = (float*)d_out;

  // workspace layout (bytes)
  char* ws = (char*)d_ws;
  _Float16* h1h = (_Float16*)ws;                               // 134,217,728 B
  float* scores = (float*)(ws + 134217728ull);                 //     524,288 B
  float* stats  = (float*)(ws + 134742016ull);                 //         128 B
  _Float16* W1h = (_Float16*)(ws + 134742144ull);              //     524,288 B
  _Float16* Wah = (_Float16*)(ws + 135266432ull);              //     262,144 B
  _Float16* Wbh = (_Float16*)(ws + 135528576ull);              //     262,144 B

  cvt_f32_f16<<<(F_DIM * H_DIM + 255) / 256, 256, 0, stream>>>(W1, W1h, F_DIM * H_DIM);
  cvt_f32_f16<<<(H_DIM * D_DIM + 255) / 256, 256, 0, stream>>>(Wa, Wah, H_DIM * D_DIM);
  cvt_f32_f16<<<(H_DIM * D_DIM + 255) / 256, 256, 0, stream>>>(Wb, Wbh, H_DIM * D_DIM);

  float* a_raw = out + 5;                 // [B,1,N] flat = 131072
  float* m_out = out + 5 + ROWS_TOTAL;    // [B,1,H] flat = 8192

  mil_phase1<<<ROWS_TOTAL / TM, 256, 0, stream>>>(
      h, W1h, b1, Wah, ba, Wbh, bb, Wc, bc, h1h, scores, a_raw);
  mil_phase2<<<B_BAGS, 256, 0, stream>>>(scores, stats);
  mil_phase3<<<B_BAGS * 4, 256, 0, stream>>>(h1h, scores, stats, m_out);
  mil_phase4<<<1, 256, 0, stream>>>(m_out, Wcls, bcls, out);
}
